// MCSF_5231270167009
// MI455X (gfx1250) — matrix operand order repaired and checked by reading, not compiled, not run on hardware
//
#include <hip/hip_runtime.h>
#include <hip/hip_bf16.h>
#include <math.h>

// ---------------- problem constants (match reference) ----------------
constexpr int kB  = 8;
constexpr int kC  = 256;
constexpr int kH  = 32;
constexpr int kW  = 32;
constexpr int kL  = kH * kW;      // 1024
constexpr int kMC = 64;           // C / RED
constexpr int kDI = 128;          // 2 * MC
constexpr int kK  = 4;            // scan directions
constexpr int kN  = 16;           // state dim
constexpr int kR  = 4;            // MC/16
constexpr int kCDBL = 36;         // R + 2N
constexpr int kCD = 48;           // padded to 3 WMMA tiles
constexpr int kMID = 64;
constexpr float kEPS = 1e-5f;

typedef __attribute__((ext_vector_type(16))) __bf16 v16bf;
typedef __attribute__((ext_vector_type(8)))  float  v8f;

// ---------------- WMMA fragment loaders (CDNA5 16x16x32 bf16 layouts) ----
// A-matrix 16x32 bf16 (ISA 7.12.2): lanes 0-15 -> M=lane, K-halves 0..7/16..23;
// lanes 16-31 -> same M, K-halves 8..15/24..31. VGPR v holds K pair.
__device__ __forceinline__ v16bf load_a_frag_rowptr(const __bf16* row) {
  const int lane  = threadIdx.x & 31;
  const int kbase = (lane & 16) ? 8 : 0;
  v16bf f;
#pragma unroll
  for (int v = 0; v < 8; ++v) {
    const int k0 = kbase + ((v < 4) ? (2 * v) : (16 + 2 * (v - 4)));
    f[2 * v]     = row[k0];
    f[2 * v + 1] = row[k0 + 1];
  }
  return f;
}
__device__ __forceinline__ v16bf load_a_frag(const __bf16* tile, int ld) {
  return load_a_frag_rowptr(tile + (size_t)(threadIdx.x & 15) * ld);
}
// B-matrix 32x16 bf16: lane -> N=lane&15; lanes 0-15 hold K=0..7/16..23,
// lanes 16-31 hold K=8..15/24..31 (same order as A).
// W is row-major [N][Ktot]; pass W + n0*ld + kk.
__device__ __forceinline__ v16bf load_b_frag(const __bf16* w, int ld) {
  const int lane = threadIdx.x & 31;
  const int n    = lane & 15;
  const int koff = (lane & 16) ? 8 : 0;
  const __bf16* p = w + (size_t)n * ld + koff;
  v16bf f;
#pragma unroll
  for (int e = 0; e < 16; ++e) f[e] = p[(e < 8) ? e : (e + 8)];
  return f;
}
#define WMMA_BF16(a, b, c) \
  __builtin_amdgcn_wmma_f32_16x16x32_bf16(false, (a), false, (b), (short)0, (c), false, false)

__device__ __forceinline__ float sigmoidf_(float x) { return 1.f / (1.f + __expf(-x)); }
__device__ __forceinline__ float softplusf_(float x) {
  return (x > 20.f) ? x : log1pf(__expf(x));
}
// direction-local index t -> row-major memory index (H=W=32)
__device__ __forceinline__ int lmap(int k, int t) {
  switch (k & 3) {
    case 0:  return t;
    case 1:  return ((t & 31) << 5) | (t >> 5);
    case 2:  return (kL - 1) - t;
    default: { int tt = (kL - 1) - t; return ((tt & 31) << 5) | (tt >> 5); }
  }
}

// ---------------- prep kernels ----------------
// rgb+dem (B,C,H,W) -> bf16 [b][l][c] (rows = pixels, cols = channels)
__global__ void k_prep_xbase(const float* __restrict__ rgb, const float* __restrict__ dem,
                             __bf16* __restrict__ xb) {
  const int idx = blockIdx.x * 256 + threadIdx.x;   // (b,c,l)
  const int l = idx & (kL - 1);
  const int c = (idx >> 10) & (kC - 1);
  const int b = idx >> 18;
  xb[(((size_t)b * kL) + l) * kC + c] = (__bf16)(rgb[idx] + dem[idx]);
}

__global__ void k_f32_to_bf16(const float* __restrict__ s, __bf16* __restrict__ d, int n) {
  const int i = blockIdx.x * 256 + threadIdx.x;
  if (i < n) d[i] = (__bf16)s[i];
}

// x_proj_w (K,36,DI) -> padded bf16 (K,48,DI)
__global__ void k_pack_xproj(const float* __restrict__ s, __bf16* __restrict__ d) {
  const int i = blockIdx.x * 256 + threadIdx.x;     // K*48*128 = 24576
  if (i >= kK * kCD * kDI) return;
  const int dd = i & (kDI - 1);
  const int c  = (i >> 7) % kCD;
  const int k  = i / (kCD * kDI);
  d[i] = (c < kCDBL) ? (__bf16)s[((size_t)k * kCDBL + c) * kDI + dd] : (__bf16)0.f;
}

// ---------------- GEMM1: channel reduce 256->64 + BatchNorm ----------------
// out x[b][l][64] f32
__global__ void k_gemm_reduce_bn(const __bf16* __restrict__ xb, const __bf16* __restrict__ wr,
                                 const float* __restrict__ g, const float* __restrict__ bt,
                                 const float* __restrict__ mu, const float* __restrict__ vr,
                                 float* __restrict__ xout) {
  const int mt = blockIdx.x, nt = blockIdx.y, b = blockIdx.z;
  const int lane = threadIdx.x;
  v8f acc = {};
  const __bf16* A0 = xb + ((size_t)b * kL + mt * 16) * kC;
#pragma unroll
  for (int kk = 0; kk < kC; kk += 32) {
    v16bf a  = load_a_frag(A0 + kk, kC);
    v16bf bb = load_b_frag(wr + (size_t)(nt * 16) * kC + kk, kC);
    acc = WMMA_BF16(a, bb, acc);
  }
  const int n  = nt * 16 + (lane & 15);
  const float sc = g[n] * rsqrtf(vr[n] + kEPS);
  const float sh = bt[n] - mu[n] * sc;
  const int mb = (lane & 16) ? 8 : 0;
#pragma unroll
  for (int r = 0; r < 8; ++r) {
    const int l = mt * 16 + mb + r;
    xout[((size_t)b * kL + l) * kMC + n] = acc[r] * sc + sh;
  }
}

// ---------------- LayerNorm over 64 -> bf16 ----------------
__global__ void k_ln64(const float* __restrict__ x, const float* __restrict__ g,
                       const float* __restrict__ bt, __bf16* __restrict__ xn) {
  const size_t row = blockIdx.x;
  const int lane = threadIdx.x;
  const float* p = x + row * kMC;
  const float v0 = p[lane], v1 = p[lane + 32];
  float s = v0 + v1, s2 = v0 * v0 + v1 * v1;
#pragma unroll
  for (int m = 16; m >= 1; m >>= 1) { s += __shfl_xor(s, m, 32); s2 += __shfl_xor(s2, m, 32); }
  const float mean = s * (1.f / 64.f);
  const float inv  = rsqrtf(s2 * (1.f / 64.f) - mean * mean + kEPS);
  __bf16* q = xn + row * kMC;
  q[lane]      = (__bf16)((v0 - mean) * inv * g[lane]      + bt[lane]);
  q[lane + 32] = (__bf16)((v1 - mean) * inv * g[lane + 32] + bt[lane + 32]);
}

// ---------------- GEMM2: in_proj 64->256, split into x1 (NCHW) and z ------
__global__ void k_gemm_inproj(const __bf16* __restrict__ xn, const __bf16* __restrict__ wp,
                              float* __restrict__ x1, float* __restrict__ z) {
  const int mt = blockIdx.x, nt = blockIdx.y, lane = threadIdx.x;
  v8f acc = {};
  const __bf16* A0 = xn + (size_t)mt * 16 * kMC;
#pragma unroll
  for (int kk = 0; kk < kMC; kk += 32) {
    v16bf a  = load_a_frag(A0 + kk, kMC);
    v16bf bb = load_b_frag(wp + (size_t)(nt * 16) * kMC + kk, kMC);
    acc = WMMA_BF16(a, bb, acc);
  }
  const int n  = nt * 16 + (lane & 15);
  const int mb = (lane & 16) ? 8 : 0;
#pragma unroll
  for (int r = 0; r < 8; ++r) {
    const int row = mt * 16 + mb + r;      // b*L + l
    const int b = row >> 10, l = row & (kL - 1);
    if (n < kDI) x1[((size_t)b * kDI + n) * kL + l] = acc[r];
    else         z[(size_t)row * kDI + (n - kDI)]   = acc[r];
  }
}

// ---------------- depthwise 3x3 SAME + bias + SiLU -> u[b][d][l] ----------
__global__ void k_dwconv(const float* __restrict__ x1, const float* __restrict__ cw,
                         const float* __restrict__ cb, float* __restrict__ u) {
  const int idx = blockIdx.x * 256 + threadIdx.x;   // (b,d,l)
  const int l = idx & (kL - 1);
  const int d = (idx >> 10) & (kDI - 1);
  const int b = idx >> 17;
  const int h = l >> 5, w = l & 31;
  const float* src = x1 + (((size_t)b * kDI + d) << 10);
  float acc = cb[d];
#pragma unroll
  for (int i = 0; i < 3; ++i) {
    const int hh = h + i - 1;
    if (hh < 0 || hh >= kH) continue;
#pragma unroll
    for (int j = 0; j < 3; ++j) {
      const int ww = w + j - 1;
      if (ww < 0 || ww >= kW) continue;
      acc += src[hh * kW + ww] * cw[d * 9 + i * 3 + j];
    }
  }
  u[idx] = acc * sigmoidf_(acc);
}

// u[b][d][l] f32 -> u_bf[b][l][d] bf16 (A-matrix source for x_proj GEMM)
__global__ void k_transpose_u(const float* __restrict__ u, __bf16* __restrict__ ub) {
  const int idx = blockIdx.x * 256 + threadIdx.x;   // (b,d,l)
  const int l = idx & (kL - 1);
  const int d = (idx >> 10) & (kDI - 1);
  const int b = idx >> 17;
  ub[(((size_t)b << 10) + l) * kDI + d] = (__bf16)u[idx];
}

// ---------------- GEMM3: x_proj (per b,k): [L x 128] x [128 x 48] ---------
// output transposed-contiguous: xdbl[b][k][t][48]
__global__ void k_gemm_xproj(const __bf16* __restrict__ ub, const __bf16* __restrict__ wp,
                             float* __restrict__ xdbl) {
  const int mt = blockIdx.x, nt = blockIdx.y;
  const int bk = blockIdx.z;
  const int b = bk >> 2, k = bk & 3;
  const int lane = threadIdx.x;
  const int gl = lmap(k, mt * 16 + (lane & 15));          // direction-mapped row
  const __bf16* row = ub + ((size_t)b * kL + gl) * kDI;
  v8f acc = {};
#pragma unroll
  for (int kk = 0; kk < kDI; kk += 32) {
    v16bf a  = load_a_frag_rowptr(row + kk);
    v16bf bb = load_b_frag(wp + (size_t)(k * kCD + nt * 16) * kDI + kk, kDI);
    acc = WMMA_BF16(a, bb, acc);
  }
  const int n  = nt * 16 + (lane & 15);
  const int mb = (lane & 16) ? 8 : 0;
  float* dst = xdbl + (size_t)bk * kL * kCD;
#pragma unroll
  for (int r = 0; r < 8; ++r) {
    const int t = mt * 16 + mb + r;
    dst[(size_t)t * kCD + n] = acc[r];
  }
}

// ---------------- selective scan: one wave = (b,k,2 channels x 16 states) --
__global__ void k_scan(const float* __restrict__ u, const float* __restrict__ xdbl,
                       const float* __restrict__ dtw, const float* __restrict__ dtb,
                       const float* __restrict__ alog, const float* __restrict__ Ds,
                       float* __restrict__ oy) {
  const int id = blockIdx.x;                 // (b*K + k)*64 + dpair
  const int dp = id & 63;
  const int bk = id >> 6;
  const int k = bk & 3, b = bk >> 2;
  const int lane = threadIdx.x;
  const int n = lane & 15;
  const int d = dp * 2 + (lane >> 4);
  const int kd = k * kDI + d;
  const float Av = -__expf(alog[(size_t)kd * kN + n]);
  const float w0 = dtw[kd * kR + 0], w1 = dtw[kd * kR + 1];
  const float w2 = dtw[kd * kR + 2], w3 = dtw[kd * kR + 3];
  const float db = dtb[kd];
  const float Dv = Ds[kd];
  const float* ubase = u + ((size_t)b * kDI + d) * kL;
  const float* xbase = xdbl + (size_t)bk * kL * kCD;
  float* oyb = oy + ((size_t)bk * kDI + d) * kL;
  float h = 0.f;
  for (int t = 0; t < kL; ++t) {
    const float* c = xbase + (size_t)t * kCD;
    const float ut  = ubase[lmap(k, t)];
    const float dts = db + c[0] * w0 + c[1] * w1 + c[2] * w2 + c[3] * w3;
    const float dlt = softplusf_(dts);
    const float Bt = c[4 + n], Ct = c[20 + n];
    h = h * __expf(dlt * Av) + (dlt * ut) * Bt;
    float p = h * Ct;
    p += __shfl_xor(p, 1, 32); p += __shfl_xor(p, 2, 32);
    p += __shfl_xor(p, 4, 32); p += __shfl_xor(p, 8, 32);
    if (n == 0) oyb[t] = p + Dv * ut;
  }
}

// ---------------- combine 4 directions -> yc[b][l][d] ----------------
__global__ void k_combine(const float* __restrict__ oy, float* __restrict__ yc) {
  const int idx = blockIdx.x * 256 + threadIdx.x;   // (b,l,d)
  const int d = idx & (kDI - 1);
  const int l = (idx >> 7) & (kL - 1);
  const int b = idx >> 17;
  const int lt = ((l & 31) << 5) | (l >> 5);
  const size_t base = (size_t)b * kK * kDI * kL + (size_t)d * kL;
  const size_t dl = (size_t)kDI * kL;
  const float y = oy[base + 0 * dl + l]
                + oy[base + 2 * dl + (kL - 1 - l)]
                + oy[base + 1 * dl + lt]
                + oy[base + 3 * dl + (kL - 1 - lt)];
  yc[idx] = y;
}

// ---------------- LN(128) + gate by silu(z) -> bf16 ----------------
__global__ void k_lngate(const float* __restrict__ yc, const float* __restrict__ z,
                         const float* __restrict__ g, const float* __restrict__ bt,
                         __bf16* __restrict__ yg) {
  const size_t row = blockIdx.x;
  const int lane = threadIdx.x;
  const float* p = yc + row * kDI;
  float v[4], s = 0.f, s2 = 0.f;
#pragma unroll
  for (int i = 0; i < 4; ++i) { v[i] = p[lane + 32 * i]; s += v[i]; s2 += v[i] * v[i]; }
#pragma unroll
  for (int m = 16; m >= 1; m >>= 1) { s += __shfl_xor(s, m, 32); s2 += __shfl_xor(s2, m, 32); }
  const float mean = s * (1.f / 128.f);
  const float inv  = rsqrtf(s2 * (1.f / 128.f) - mean * mean + kEPS);
#pragma unroll
  for (int i = 0; i < 4; ++i) {
    const int c = lane + 32 * i;
    const float ln = (v[i] - mean) * inv * g[c] + bt[c];
    const float zv = z[row * kDI + c];
    yg[row * kDI + c] = (__bf16)(ln * zv * sigmoidf_(zv));
  }
}

// ---------------- GEMM4: out_proj 128->64, + residual x -> s ---------------
__global__ void k_gemm_outproj(const __bf16* __restrict__ yg, const __bf16* __restrict__ wp,
                               const float* __restrict__ x, float* __restrict__ sout) {
  const int mt = blockIdx.x, nt = blockIdx.y, lane = threadIdx.x;
  v8f acc = {};
  const __bf16* A0 = yg + (size_t)mt * 16 * kDI;
#pragma unroll
  for (int kk = 0; kk < kDI; kk += 32) {
    v16bf a  = load_a_frag(A0 + kk, kDI);
    v16bf bb = load_b_frag(wp + (size_t)(nt * 16) * kDI + kk, kDI);
    acc = WMMA_BF16(a, bb, acc);
  }
  const int n  = nt * 16 + (lane & 15);
  const int mb = (lane & 16) ? 8 : 0;
#pragma unroll
  for (int r = 0; r < 8; ++r) {
    const size_t row = (size_t)mt * 16 + mb + r;
    sout[row * kMC + n] = acc[r] + x[row * kMC + n];
  }
}

// ---------------- LN(mno) over 64 -> xm[b][l][64] ----------------
__global__ void k_ln_mno(const float* __restrict__ s, const float* __restrict__ g,
                         const float* __restrict__ bt, float* __restrict__ xm) {
  const size_t row = blockIdx.x;
  const int lane = threadIdx.x;
  const float* p = s + row * kMC;
  const float v0 = p[lane], v1 = p[lane + 32];
  float ss = v0 + v1, s2 = v0 * v0 + v1 * v1;
#pragma unroll
  for (int m = 16; m >= 1; m >>= 1) { ss += __shfl_xor(ss, m, 32); s2 += __shfl_xor(s2, m, 32); }
  const float mean = ss * (1.f / 64.f);
  const float inv  = rsqrtf(s2 * (1.f / 64.f) - mean * mean + kEPS);
  float* q = xm + row * kMC;
  q[lane]      = (v0 - mean) * inv * g[lane]      + bt[lane];
  q[lane + 32] = (v1 - mean) * inv * g[lane + 32] + bt[lane + 32];
}

// deterministic spatial mean of xm -> zm[b][64]
__global__ void k_pool_zm(const float* __restrict__ xm, float* __restrict__ zm) {
  const int id = blockIdx.x;          // b*64 + c
  const int b = id >> 6, c = id & 63;
  const float* p = xm + ((size_t)b << 10) * kMC + c;
  float s = 0.f;
  for (int l = threadIdx.x; l < kL; l += 32) s += p[(size_t)l * kMC];
#pragma unroll
  for (int m = 16; m >= 1; m >>= 1) s += __shfl_xor(s, m, 32);
  if (threadIdx.x == 0) zm[id] = s * (1.f / (float)kL);
}

// spatial means of rgb / dem
__global__ void k_pool_rgbdem(const float* __restrict__ rgb, const float* __restrict__ dem,
                              float* __restrict__ zr, float* __restrict__ zd) {
  const int id = blockIdx.x;          // 0..2*B*C-1
  const int sel = id >> 11;
  const int rem = id & 2047;
  const float* src = (sel ? dem : rgb) + ((size_t)rem << 10);
  float s = 0.f;
  for (int l = threadIdx.x; l < kL; l += 32) s += src[l];
#pragma unroll
  for (int m = 16; m >= 1; m >>= 1) s += __shfl_xor(s, m, 32);
  if (threadIdx.x == 0) (sel ? zd : zr)[rem] = s * (1.f / (float)kL);
}

// ---------------- joint-embedding MLP (tiny): 576 -> 64 -> 512 -------------
__global__ void k_je(const float* __restrict__ zr, const float* __restrict__ zd,
                     const float* __restrict__ zm, const float* __restrict__ w1,
                     const float* __restrict__ g1, const float* __restrict__ b1,
                     const float* __restrict__ m1, const float* __restrict__ v1,
                     const float* __restrict__ w2, float* __restrict__ wgt) {
  __shared__ float zj[2 * kC + kMC];
  __shared__ float hm[kMID];
  const int b = blockIdx.x, t = threadIdx.x;
  for (int i = t; i < kC; i += 64) {
    zj[i]      = zr[(size_t)b * kC + i];
    zj[kC + i] = zd[(size_t)b * kC + i];
  }
  if (t < kMC) zj[2 * kC + t] = zm[(size_t)b * kMC + t];
  __syncthreads();
  // hidden
  float a = 0.f;
  const float* wr = w1 + (size_t)t * (2 * kC + kMC);
  for (int i = 0; i < 2 * kC + kMC; ++i) a += zj[i] * wr[i];
  a = (a - m1[t]) * rsqrtf(v1[t] + kEPS) * g1[t] + b1[t];
  hm[t] = fmaxf(a, 0.f);
  __syncthreads();
  for (int o = t; o < 2 * kC; o += 64) {
    float s = 0.f;
    const float* w = w2 + (size_t)o * kMID;
    for (int j = 0; j < kMID; ++j) s += hm[j] * w[j];
    wgt[(size_t)b * 2 * kC + o] = sigmoidf_(s);
  }
}

// ---------------- final reweight ----------------
__global__ void k_final(const float* __restrict__ rgb, const float* __restrict__ dem,
                        const float* __restrict__ wgt, float* __restrict__ out) {
  const int idx = blockIdx.x * 256 + threadIdx.x;   // (b,c,l)
  const int c = (idx >> 10) & (kC - 1);
  const int b = idx >> 18;
  const float wr = wgt[(size_t)b * 2 * kC + c];
  const float wd = wgt[(size_t)b * 2 * kC + kC + c];
  out[idx] = rgb[idx] * (1.f + wr) + dem[idx] * (1.f + wd);
}

// ---------------- host launcher ----------------
static inline size_t alignup(size_t x) { return (x + 255) & ~(size_t)255; }

extern "C" void kernel_launch(void* const* d_in, const int* in_sizes, int n_in,
                              void* d_out, int out_size, void* d_ws, size_t ws_size,
                              hipStream_t stream) {
  (void)in_sizes; (void)n_in; (void)out_size; (void)ws_size;
  const float* rgb   = (const float*)d_in[0];
  const float* dem   = (const float*)d_in[1];
  const float* w_red = (const float*)d_in[2];
  const float* bnr_g = (const float*)d_in[3];
  const float* bnr_b = (const float*)d_in[4];
  const float* bnr_m = (const float*)d_in[5];
  const float* bnr_v = (const float*)d_in[6];
  const float* ln1_g = (const float*)d_in[7];
  const float* ln1_b = (const float*)d_in[8];
  const float* inp_w = (const float*)d_in[9];
  const float* conv_w = (const float*)d_in[10];
  const float* conv_b = (const float*)d_in[11];
  const float* xproj_w = (const float*)d_in[12];
  const float* dt_w  = (const float*)d_in[13];
  const float* dt_b  = (const float*)d_in[14];
  const float* A_logs = (const float*)d_in[15];
  const float* Ds    = (const float*)d_in[16];
  const float* on_g  = (const float*)d_in[17];
  const float* on_b  = (const float*)d_in[18];
  const float* outp_w = (const float*)d_in[19];
  const float* mno_g = (const float*)d_in[20];
  const float* mno_b = (const float*)d_in[21];
  const float* je_w1 = (const float*)d_in[22];
  const float* bn1_g = (const float*)d_in[23];
  const float* bn1_b = (const float*)d_in[24];
  const float* bn1_m = (const float*)d_in[25];
  const float* bn1_v = (const float*)d_in[26];
  const float* je_w2 = (const float*)d_in[27];
  float* out = (float*)d_out;

  // ---- workspace carve-up ----
  char* ws = (char*)d_ws;
  size_t off = 0;
  auto take = [&](size_t bytes) { char* p = ws + off; off += alignup(bytes); return p; };
  __bf16* xb_bf  = (__bf16*)take((size_t)kB * kL * kC * 2);
  __bf16* wr_bf  = (__bf16*)take((size_t)kMC * kC * 2);
  __bf16* ip_bf  = (__bf16*)take((size_t)2 * kDI * kMC * 2);
  __bf16* xp_bf  = (__bf16*)take((size_t)kK * kCD * kDI * 2);
  __bf16* op_bf  = (__bf16*)take((size_t)kMC * kDI * 2);
  float*  x_f    = (float*)take((size_t)kB * kL * kMC * 4);
  __bf16* xn_bf  = (__bf16*)take((size_t)kB * kL * kMC * 2);
  float*  x1_f   = (float*)take((size_t)kB * kDI * kL * 4);
  float*  z_f    = (float*)take((size_t)kB * kL * kDI * 4);
  float*  u_f    = (float*)take((size_t)kB * kDI * kL * 4);
  __bf16* u_bf   = (__bf16*)take((size_t)kB * kL * kDI * 2);
  float*  xdbl   = (float*)take((size_t)kB * kK * kL * kCD * 4);
  float*  oy     = (float*)take((size_t)kB * kK * kDI * kL * 4);
  float*  yc     = (float*)take((size_t)kB * kL * kDI * 4);
  __bf16* yg_bf  = (__bf16*)take((size_t)kB * kL * kDI * 2);
  float*  s_f    = (float*)take((size_t)kB * kL * kMC * 4);
  float*  xm_f   = (float*)take((size_t)kB * kL * kMC * 4);
  float*  zm     = (float*)take((size_t)kB * kMC * 4);
  float*  zr     = (float*)take((size_t)kB * kC * 4);
  float*  zd     = (float*)take((size_t)kB * kC * 4);
  float*  wgt    = (float*)take((size_t)kB * 2 * kC * 4);

  // ---- pipeline ----
  k_prep_xbase<<<(kB * kC * kL) / 256, 256, 0, stream>>>(rgb, dem, xb_bf);
  k_f32_to_bf16<<<(kMC * kC + 255) / 256, 256, 0, stream>>>(w_red, wr_bf, kMC * kC);
  k_f32_to_bf16<<<(2 * kDI * kMC + 255) / 256, 256, 0, stream>>>(inp_w, ip_bf, 2 * kDI * kMC);
  k_f32_to_bf16<<<(kMC * kDI + 255) / 256, 256, 0, stream>>>(outp_w, op_bf, kMC * kDI);
  k_pack_xproj<<<(kK * kCD * kDI + 255) / 256, 256, 0, stream>>>(xproj_w, xp_bf);

  k_gemm_reduce_bn<<<dim3(kL / 16, kMC / 16, kB), 32, 0, stream>>>(
      xb_bf, wr_bf, bnr_g, bnr_b, bnr_m, bnr_v, x_f);
  k_ln64<<<kB * kL, 32, 0, stream>>>(x_f, ln1_g, ln1_b, xn_bf);
  k_gemm_inproj<<<dim3(kB * kL / 16, (2 * kDI) / 16), 32, 0, stream>>>(xn_bf, ip_bf, x1_f, z_f);
  k_dwconv<<<(kB * kDI * kL) / 256, 256, 0, stream>>>(x1_f, conv_w, conv_b, u_f);
  k_transpose_u<<<(kB * kDI * kL) / 256, 256, 0, stream>>>(u_f, u_bf);
  k_gemm_xproj<<<dim3(kL / 16, kCD / 16, kB * kK), 32, 0, stream>>>(u_bf, xp_bf, xdbl);
  k_scan<<<kB * kK * (kDI / 2), 32, 0, stream>>>(u_f, xdbl, dt_w, dt_b, A_logs, Ds, oy);
  k_combine<<<(kB * kL * kDI) / 256, 256, 0, stream>>>(oy, yc);
  k_lngate<<<kB * kL, 32, 0, stream>>>(yc, z_f, on_g, on_b, yg_bf);
  k_gemm_outproj<<<dim3(kB * kL / 16, kMC / 16), 32, 0, stream>>>(yg_bf, op_bf, x_f, s_f);
  k_ln_mno<<<kB * kL, 32, 0, stream>>>(s_f, mno_g, mno_b, xm_f);
  k_pool_zm<<<kB * kMC, 32, 0, stream>>>(xm_f, zm);
  k_pool_rgbdem<<<2 * kB * kC, 32, 0, stream>>>(rgb, dem, zr, zd);
  k_je<<<kB, 64, 0, stream>>>(zr, zd, zm, je_w1, bn1_g, bn1_b, bn1_m, bn1_v, je_w2, wgt);
  k_final<<<(kB * kC * kL) / 256, 256, 0, stream>>>(rgb, dem, wgt, out);
}
